// SwinTransformerBlock3D_36318243455750
// MI455X (gfx1250) — matrix operand order repaired and checked by reading, not compiled, not run on hardware
//
#include <hip/hip_runtime.h>
#include <hip/hip_bf16.h>

// ---------------------------------------------------------------------------
// Swin-3D block for MI455X (gfx1250, wave32).  All GEMMs use
// v_wmma_f32_16x16x32_f16 (f16 A/B, f32 accumulate).
// ---------------------------------------------------------------------------

typedef __attribute__((ext_vector_type(16))) _Float16 v16h;
typedef __attribute__((ext_vector_type(8)))  _Float16 v8h;
typedef __attribute__((ext_vector_type(8)))  float    v8f;

#define NTOK   200704      // 8*8*56*56 tokens = 2048 windows * 98
#define NWIN   2048
#define NWH    6144        // windows * heads
#define GEMM_BLOCKS 3136   // (NTOK/16)/4 strips per block

__device__ __forceinline__ v16h join16(v8h lo, v8h hi) {
  return __builtin_shufflevector(lo, hi, 0,1,2,3,4,5,6,7,8,9,10,11,12,13,14,15);
}

// A fragment (16x32, f16). A row-major [M][lda]. Per ISA: lane l holds row
// m0+(l&15); element i holds K = k0 + 16*(i>>3) + 8*(l>>4) + (i&7).
__device__ __forceinline__ v16h load_a(const _Float16* A, int lda, int m0, int k0, int lane) {
  const _Float16* p = A + (size_t)(m0 + (lane & 15)) * lda + k0 + ((lane >> 4) << 3);
  v8h lo = *(const v8h*)p;          // K = k0 + 8*hi + [0..7]
  v8h hi = *(const v8h*)(p + 16);   // K = k0 + 16 + 8*hi + [0..7]
  return join16(lo, hi);
}

// B fragment (32x16, f16), column-major per lane: BT[N][ldk] where BT row n is
// B's column n. Lane l holds column n0+(l&15); element i holds K = k0+16*(i>>3)+8*(l>>4)+(i&7).
__device__ __forceinline__ v16h load_b(const _Float16* BT, int ldk, int n0, int k0, int lane) {
  const _Float16* p = BT + (size_t)(n0 + (lane & 15)) * ldk + k0 + ((lane >> 4) << 3);
  v8h lo = *(const v8h*)p;
  v8h hi = *(const v8h*)(p + 16);
  return join16(lo, hi);
}

__device__ __forceinline__ v8f wmma16(v16h a, v16h b, v8f c) {
  return __builtin_amdgcn_wmma_f32_16x16x32_f16(false, a, false, b, (short)0, c, false, false);
}

// ---------------------------------------------------------------------------
// prep kernels
// ---------------------------------------------------------------------------
__global__ void cvt_f16_kernel(const float* __restrict__ s, _Float16* __restrict__ d, int n) {
  int i = blockIdx.x * blockDim.x + threadIdx.x;
  if (i < n) d[i] = (_Float16)s[i];
}

// Padded bias table biasPad[h][112][112]:
//   m >= 98            -> -1e30   (softmax column mask folded into the table)
//   n >= 98 (m < 98)   ->  0      (pad rows, finite; results discarded)
//   else               -> rpb_table[REL_IDX[n][m]*3 + h]
__global__ void build_bias_kernel(const float* __restrict__ rpb, float* __restrict__ bias) {
  int i = blockIdx.x * blockDim.x + threadIdx.x;
  if (i >= 3 * 112 * 112) return;
  int m = i % 112, n = (i / 112) % 112, h = i / (112 * 112);
  float v;
  if (m >= 98) {
    v = -1e30f;
  } else if (n >= 98) {
    v = 0.f;
  } else {
    int ddn = n / 49, hhn = (n / 7) % 7, wwn = n % 7;
    int ddm = m / 49, hhm = (m / 7) % 7, wwm = m % 7;
    int idx = (ddn - ddm + 1) * 169 + (hhn - hhm + 6) * 13 + (wwn - wwm + 6);
    v = rpb[idx * 3 + h];
  }
  bias[i] = v;
}

// zero padding rows 98..127 of vT (P columns there are exact zero, but the
// padded v values must be finite; ws contents are untrusted)
__global__ void zero_vpad_kernel(_Float16* __restrict__ vT) {
  int i = blockIdx.x * blockDim.x + threadIdx.x;
  const int total = NWH * 32 * 30;
  if (i >= total) return;
  int m = 98 + (i % 30);
  int rest = i / 30;                       // (wh*32 + d)
  vT[(size_t)rest * 128 + m] = (_Float16)0.f;
}

// ---------------------------------------------------------------------------
// LayerNorm1 + window partition:  x (natural) -> xw (window order, f16)
// one wave per token; 96 channels = 3 per lane
// ---------------------------------------------------------------------------
__global__ __launch_bounds__(128) void ln1_part_kernel(
    const float* __restrict__ x, const float* __restrict__ w, const float* __restrict__ b,
    _Float16* __restrict__ xw) {
  int tok  = blockIdx.x * 4 + (threadIdx.x >> 5);
  int lane = threadIdx.x & 31;
  const float* px = x + (size_t)tok * 96;
  float v0 = px[lane], v1 = px[lane + 32], v2 = px[lane + 64];
  float s = v0 + v1 + v2;
  for (int o = 16; o; o >>= 1) s += __shfl_xor(s, o, 32);
  float mean = s * (1.f / 96.f);
  float d0 = v0 - mean, d1 = v1 - mean, d2 = v2 - mean;
  float q = d0 * d0 + d1 * d1 + d2 * d2;
  for (int o = 16; o; o >>= 1) q += __shfl_xor(q, o, 32);
  float r = rsqrtf(q * (1.f / 96.f) + 1e-5f);
  // natural token -> (window, token-in-window)
  int bb = tok / 25088;      int rr  = tok % 25088;
  int d  = rr / 3136;        int rem = rr % 3136;
  int hp = rem / 56;         int wp  = rem % 56;
  int wN = ((bb * 4 + (d >> 1)) * 8 + hp / 7) * 8 + wp / 7;
  int nN = ((d & 1) * 7 + hp % 7) * 7 + wp % 7;
  _Float16* pd = xw + ((size_t)wN * 98 + nN) * 96;
  pd[lane]      = (_Float16)(d0 * r * w[lane]      + b[lane]);
  pd[lane + 32] = (_Float16)(d1 * r * w[lane + 32] + b[lane + 32]);
  pd[lane + 64] = (_Float16)(d2 * r * w[lane + 64] + b[lane + 64]);
}

// ---------------------------------------------------------------------------
// QKV GEMM: xw(200704x96) @ qkv_w^T(96x288) + b ; scatter to q/k/vT per head.
// B fragments hoisted ahead of the WMMA chain so loads clause together and
// the waits stagger.  part/head/dch-base are wave-uniform per N-tile.
// q,k: [wh][112][32] f16 ; vT: [wh][32][128] f16 (transposed for P@V B-frags)
// ---------------------------------------------------------------------------
__global__ __launch_bounds__(128) void qkv_gemm_kernel(
    const _Float16* __restrict__ xw, const _Float16* __restrict__ wq,
    const float* __restrict__ qb_bias,
    _Float16* __restrict__ q, _Float16* __restrict__ k, _Float16* __restrict__ vT) {
  int lane = threadIdx.x & 31;
  int m0   = (blockIdx.x * 4 + (threadIdx.x >> 5)) * 16;
  int hi = lane >> 4, l15 = lane & 15;
  v16h a0 = load_a(xw, 96, m0,  0, lane);
  v16h a1 = load_a(xw, 96, m0, 32, lane);
  v16h a2 = load_a(xw, 96, m0, 64, lane);
  // per-wave row decode (shared by all N-tiles)
  int wN8[8], nn8[8];
  #pragma unroll
  for (int j = 0; j < 8; ++j) {
    int m = m0 + j + hi * 8;
    wN8[j] = m / 98;
    nn8[j] = m - wN8[j] * 98;
  }
  #pragma unroll
  for (int nt = 0; nt < 18; ++nt) {
    const int n0   = nt * 16;
    const int part = n0 / 96;               // 0=q 1=k 2=v   (uniform)
    const int hh   = (n0 % 96) >> 5;        // head          (uniform)
    const int dch  = (n0 & 31) + l15;       // head channel  (per-lane)
    v16h b0 = load_b(wq, 96, n0,  0, lane);
    v16h b1 = load_b(wq, 96, n0, 32, lane);
    v16h b2 = load_b(wq, 96, n0, 64, lane);
    v8f c = {};
    c = wmma16(a0, b0, c);
    c = wmma16(a1, b1, c);
    c = wmma16(a2, b2, c);
    float bv = qb_bias[n0 + l15];
    #pragma unroll
    for (int j = 0; j < 8; ++j) {
      size_t base = (size_t)wN8[j] * 3 + hh;
      float v = c[j] + bv;
      if (part == 0)       q [(base * 112 + nn8[j]) * 32 + dch] = (_Float16)(v * 0.17677669529663687f);
      else if (part == 1)  k [(base * 112 + nn8[j]) * 32 + dch] = (_Float16)v;
      else                 vT[(base * 32 + dch) * 128 + nn8[j]] = (_Float16)v;
    }
  }
}

// ---------------------------------------------------------------------------
// Attention per (window, head).  S=q@k^T+biasPad (WMMA -> LDS, branchless),
// softmax rows (in-place f32 -> f16), O=P@v (WMMA from LDS).
// ---------------------------------------------------------------------------
__global__ __launch_bounds__(128) void attn_kernel(
    const _Float16* __restrict__ qg, const _Float16* __restrict__ kg,
    const _Float16* __restrict__ vTg, const float* __restrict__ bias,
    _Float16* __restrict__ out) {
  __shared__ float S[112 * 128];                       // 57,344 B
  int wh = blockIdx.x;
  int h = wh % 3, wN = wh / 3;
  int wave = threadIdx.x >> 5, lane = threadIdx.x & 31;
  int hi = lane >> 4, l15 = lane & 15;
  const _Float16* qb = qg  + (size_t)wh * 112 * 32;
  const _Float16* kb = kg  + (size_t)wh * 112 * 32;
  const _Float16* vb = vTg + (size_t)wh * 32 * 128;
  const float*    bh = bias + (size_t)h * 112 * 112;

  // Phase 1: S tiles (7x7 tiles of 16x16), K=32 -> single WMMA per tile
  for (int st = wave; st < 7; st += 4) {
    v16h a = load_a(qb, 32, st * 16, 0, lane);
    for (int mt = 0; mt < 7; ++mt) {
      v16h bfr = load_b(kb, 32, mt * 16, 0, lane);
      v8f c = {};
      c = wmma16(a, bfr, c);
      int mcol = mt * 16 + l15;
      int nrow = st * 16 + hi * 8;
      #pragma unroll
      for (int j = 0; j < 8; ++j) {
        int n = nrow + j;
        S[n * 128 + mcol] = c[j] + bh[n * 112 + mcol];   // mask folded in table
      }
    }
  }
  __syncthreads();

  // Phase 2: row softmax; convert in place to f16 P rows (stride 256 halves)
  int row = threadIdx.x;
  if (row < 112) {
    float* sr = &S[row * 128];
    float mx = -1e30f;
    for (int m = 0; m < 98; ++m) mx = fmaxf(mx, sr[m]);
    float sum = 0.f;
    for (int m = 0; m < 98; ++m) sum += __expf(sr[m] - mx);
    float inv = 1.f / sum;
    _Float16* ph = (_Float16*)sr;
    for (int m = 0; m < 98; ++m) {
      float e = __expf(sr[m] - mx) * inv;
      asm volatile("" ::: "memory");   // order f32 load vs f16 in-place store
      ph[m] = (_Float16)e;
    }
    for (int m = 98; m < 128; ++m) ph[m] = (_Float16)0.f;
  }
  __syncthreads();

  // Phase 3: O = P @ v   (K padded to 128 -> 4 WMMA k-tiles)
  const _Float16* P = (const _Float16*)S;              // lda = 256 halves
  for (int st = wave; st < 7; st += 4) {
    v16h a0 = load_a(P, 256, st * 16,  0, lane);
    v16h a1 = load_a(P, 256, st * 16, 32, lane);
    v16h a2 = load_a(P, 256, st * 16, 64, lane);
    v16h a3 = load_a(P, 256, st * 16, 96, lane);
    for (int dt = 0; dt < 2; ++dt) {
      v16h b0 = load_b(vb, 128, dt * 16,  0, lane);
      v16h b1 = load_b(vb, 128, dt * 16, 32, lane);
      v16h b2 = load_b(vb, 128, dt * 16, 64, lane);
      v16h b3 = load_b(vb, 128, dt * 16, 96, lane);
      v8f c = {};
      c = wmma16(a0, b0, c);
      c = wmma16(a1, b1, c);
      c = wmma16(a2, b2, c);
      c = wmma16(a3, b3, c);
      int d = dt * 16 + l15;
      #pragma unroll
      for (int j = 0; j < 8; ++j) {
        int n = st * 16 + j + hi * 8;
        if (n < 98)
          out[((size_t)wN * 98 + n) * 96 + h * 32 + d] = (_Float16)c[j];
      }
    }
  }
}

// ---------------------------------------------------------------------------
// window token index -> natural flat token index (window_reverse)
// ---------------------------------------------------------------------------
__device__ __forceinline__ int nat_idx(int t) {
  int wN = t / 98, n = t - wN * 98;
  int bb = wN >> 8, r = wN & 255;
  int d2 = r >> 6, r2 = r & 63, h7 = r2 >> 3, w7 = r2 & 7;
  int dd = n / 49, r3 = n % 49, hh = r3 / 7, ww = r3 % 7;
  int d  = d2 * 2 + dd, hp = h7 * 7 + hh, wp = w7 * 7 + ww;
  return ((bb * 8 + d) * 56 + hp) * 56 + wp;
}

// proj GEMM + window_reverse + residual -> x2 (f32, natural layout)
__global__ __launch_bounds__(128) void proj_res_kernel(
    const _Float16* __restrict__ ao, const _Float16* __restrict__ wpw,
    const float* __restrict__ pb, const float* __restrict__ x, float* __restrict__ x2) {
  int lane = threadIdx.x & 31;
  int m0   = (blockIdx.x * 4 + (threadIdx.x >> 5)) * 16;
  int hi = lane >> 4, l15 = lane & 15;
  v16h a0 = load_a(ao, 96, m0,  0, lane);
  v16h a1 = load_a(ao, 96, m0, 32, lane);
  v16h a2 = load_a(ao, 96, m0, 64, lane);
  int gidx[8];
  #pragma unroll
  for (int j = 0; j < 8; ++j) gidx[j] = nat_idx(m0 + j + hi * 8);
  #pragma unroll
  for (int nt = 0; nt < 6; ++nt) {
    int n0 = nt * 16;
    v16h b0 = load_b(wpw, 96, n0,  0, lane);
    v16h b1 = load_b(wpw, 96, n0, 32, lane);
    v16h b2 = load_b(wpw, 96, n0, 64, lane);
    v8f c = {};
    c = wmma16(a0, b0, c);
    c = wmma16(a1, b1, c);
    c = wmma16(a2, b2, c);
    int col = n0 + l15;
    float bv = pb[col];
    #pragma unroll
    for (int j = 0; j < 8; ++j) {
      size_t g = (size_t)gidx[j] * 96 + col;
      x2[g] = x[g] + c[j] + bv;
    }
  }
}

// LayerNorm2 (natural layout), f32 -> f16
__global__ __launch_bounds__(128) void ln2_kernel(
    const float* __restrict__ x2, const float* __restrict__ w, const float* __restrict__ b,
    _Float16* __restrict__ xn) {
  int tok  = blockIdx.x * 4 + (threadIdx.x >> 5);
  int lane = threadIdx.x & 31;
  const float* px = x2 + (size_t)tok * 96;
  float v0 = px[lane], v1 = px[lane + 32], v2 = px[lane + 64];
  float s = v0 + v1 + v2;
  for (int o = 16; o; o >>= 1) s += __shfl_xor(s, o, 32);
  float mean = s * (1.f / 96.f);
  float d0 = v0 - mean, d1 = v1 - mean, d2 = v2 - mean;
  float q = d0 * d0 + d1 * d1 + d2 * d2;
  for (int o = 16; o; o >>= 1) q += __shfl_xor(q, o, 32);
  float r = rsqrtf(q * (1.f / 96.f) + 1e-5f);
  _Float16* pd = xn + (size_t)tok * 96;
  pd[lane]      = (_Float16)(d0 * r * w[lane]      + b[lane]);
  pd[lane + 32] = (_Float16)(d1 * r * w[lane + 32] + b[lane + 32]);
  pd[lane + 64] = (_Float16)(d2 * r * w[lane + 64] + b[lane + 64]);
}

// FC1 + exact GELU -> h (f16, 200704x384)
__global__ __launch_bounds__(128) void fc1_gelu_kernel(
    const _Float16* __restrict__ xn, const _Float16* __restrict__ w1,
    const float* __restrict__ b1, _Float16* __restrict__ hbuf) {
  int lane = threadIdx.x & 31;
  int m0   = (blockIdx.x * 4 + (threadIdx.x >> 5)) * 16;
  int hi = lane >> 4, l15 = lane & 15;
  v16h a0 = load_a(xn, 96, m0,  0, lane);
  v16h a1 = load_a(xn, 96, m0, 32, lane);
  v16h a2 = load_a(xn, 96, m0, 64, lane);
  for (int nt = 0; nt < 24; ++nt) {
    int n0 = nt * 16;
    v16h b0 = load_b(w1, 96, n0,  0, lane);
    v16h b1 = load_b(w1, 96, n0, 32, lane);
    v16h b2 = load_b(w1, 96, n0, 64, lane);
    v8f c = {};
    c = wmma16(a0, b0, c);
    c = wmma16(a1, b1, c);
    c = wmma16(a2, b2, c);
    int col = n0 + l15;
    float bv = b1[col];
    #pragma unroll
    for (int j = 0; j < 8; ++j) {
      float v = c[j] + bv;
      v = 0.5f * v * (1.f + erff(v * 0.70710678118654752f));
      hbuf[(size_t)(m0 + j + hi * 8) * 384 + col] = (_Float16)v;
    }
  }
}

// FC2 + residual -> out (f32, natural layout)
__global__ __launch_bounds__(128) void fc2_res_kernel(
    const _Float16* __restrict__ hbuf, const _Float16* __restrict__ w2,
    const float* __restrict__ b2, const float* __restrict__ x2, float* __restrict__ out) {
  int lane = threadIdx.x & 31;
  int m0   = (blockIdx.x * 4 + (threadIdx.x >> 5)) * 16;
  int hi = lane >> 4, l15 = lane & 15;
  v8f zero = {};
  v8f acc[6];
  #pragma unroll
  for (int i = 0; i < 6; ++i) acc[i] = zero;
  for (int kb = 0; kb < 3; ++kb) {
    v16h a[4];
    #pragma unroll
    for (int kk = 0; kk < 4; ++kk) a[kk] = load_a(hbuf, 384, m0, (kb * 4 + kk) * 32, lane);
    #pragma unroll
    for (int nt = 0; nt < 6; ++nt) {
      v16h b[4];
      #pragma unroll
      for (int kk = 0; kk < 4; ++kk) b[kk] = load_b(w2, 384, nt * 16, (kb * 4 + kk) * 32, lane);
      #pragma unroll
      for (int kk = 0; kk < 4; ++kk) acc[nt] = wmma16(a[kk], b[kk], acc[nt]);
    }
  }
  #pragma unroll
  for (int nt = 0; nt < 6; ++nt) {
    int col = nt * 16 + l15;
    float bv = b2[col];
    #pragma unroll
    for (int j = 0; j < 8; ++j) {
      size_t g = (size_t)(m0 + j + hi * 8) * 96 + col;
      out[g] = x2[g] + acc[nt][j] + bv;
    }
  }
}

// ---------------------------------------------------------------------------
// launch
// ---------------------------------------------------------------------------
extern "C" void kernel_launch(void* const* d_in, const int* in_sizes, int n_in,
                              void* d_out, int out_size, void* d_ws, size_t ws_size,
                              hipStream_t stream) {
  const float* x     = (const float*)d_in[0];
  const float* n1w   = (const float*)d_in[1];
  const float* n1b   = (const float*)d_in[2];
  const float* qkvw  = (const float*)d_in[3];
  const float* qkvb  = (const float*)d_in[4];
  const float* projw = (const float*)d_in[5];
  const float* projb = (const float*)d_in[6];
  const float* rpb   = (const float*)d_in[7];
  const float* n2w   = (const float*)d_in[8];
  const float* n2b   = (const float*)d_in[9];
  const float* f1w   = (const float*)d_in[10];
  const float* f1b   = (const float*)d_in[11];
  const float* f2w   = (const float*)d_in[12];
  const float* f2b   = (const float*)d_in[13];
  float* out = (float*)d_out;

  char* ws = (char*)d_ws;
  size_t off = 0;
  auto take = [&](size_t bytes) -> char* {
    char* p = ws + off;
    off = (off + bytes + 255) & ~(size_t)255;
    return p;
  };
  _Float16* wq  = (_Float16*)take((size_t)288 * 96 * 2);
  _Float16* wp  = (_Float16*)take((size_t)96 * 96 * 2);
  _Float16* w1  = (_Float16*)take((size_t)384 * 96 * 2);
  _Float16* w2  = (_Float16*)take((size_t)96 * 384 * 2);
  float*    bias= (float*)   take((size_t)3 * 112 * 112 * 4);
  _Float16* xw  = (_Float16*)take((size_t)NTOK * 96 * 2);        // reused as attn_out
  _Float16* qb  = (_Float16*)take((size_t)NWH * 112 * 32 * 2);   // reused as xn2
  _Float16* kb  = (_Float16*)take((size_t)NWH * 112 * 32 * 2);
  _Float16* vT  = (_Float16*)take((size_t)NWH * 32 * 128 * 2);
  float*    x2  = (float*)   take((size_t)NTOK * 96 * 4);
  _Float16* hbuf= (_Float16*)take((size_t)NTOK * 384 * 2);
  _Float16* attn_out = xw;   // xw dead after QKV
  _Float16* xn2      = qb;   // q dead after attention

  // weight conversion + padded bias table
  cvt_f16_kernel<<<(288 * 96 + 255) / 256, 256, 0, stream>>>(qkvw, wq, 288 * 96);
  cvt_f16_kernel<<<(96 * 96 + 255) / 256, 256, 0, stream>>>(projw, wp, 96 * 96);
  cvt_f16_kernel<<<(384 * 96 + 255) / 256, 256, 0, stream>>>(f1w, w1, 384 * 96);
  cvt_f16_kernel<<<(96 * 384 + 255) / 256, 256, 0, stream>>>(f2w, w2, 96 * 384);
  build_bias_kernel<<<(3 * 112 * 112 + 255) / 256, 256, 0, stream>>>(rpb, bias);

  // LN1 + window partition
  ln1_part_kernel<<<NTOK / 4, 128, 0, stream>>>(x, n1w, n1b, xw);

  // QKV GEMM + head split
  qkv_gemm_kernel<<<GEMM_BLOCKS, 128, 0, stream>>>(xw, wq, qkvb, qb, kb, vT);
  zero_vpad_kernel<<<(NWH * 32 * 30 + 255) / 256, 256, 0, stream>>>(vT);

  // attention per (window, head)
  attn_kernel<<<NWH, 128, 0, stream>>>(qb, kb, vT, bias, attn_out);

  // proj + window reverse + residual
  proj_res_kernel<<<GEMM_BLOCKS, 128, 0, stream>>>(attn_out, wp, projb, x, x2);

  // MLP
  ln2_kernel<<<NTOK / 4, 128, 0, stream>>>(x2, n2w, n2b, xn2);
  fc1_gelu_kernel<<<GEMM_BLOCKS, 128, 0, stream>>>(xn2, w1, f1b, hbuf);
  fc2_res_kernel<<<GEMM_BLOCKS, 128, 0, stream>>>(hbuf, w2, f2b, x2, out);
}
